// Generator_72756745994339
// MI455X (gfx1250) — compile-verified, matrix operand order checked by reading, not run on hardware
//
#include <hip/hip_runtime.h>
#include <hip/hip_bf16.h>

// ---------------------------------------------------------------------------
// Seq2seq (encoder LSTM + attention decoder) for MI455X / gfx1250.
//
// - Hoist time-invariant GEMMs (x@enc_Wih.T for all 128 steps;
//   noise1@emb_W_noise.T for all 64 steps) into two large WMMA GEMMs with
//   on-the-fly fp32->bf16 conversion and 4x M-register-blocking (one B
//   fragment feeds 4 WMMAs -> 4x less re-streaming of the 488MB weight).
// - Convert all *reused* weights to padded bf16 once (halves the dominant
//   weight-bandwidth cost of the 64 sequential decoder steps; emb_W/out_W
//   bf16 copies individually fit the 192MB L2).
// - Sequential recurrences run as small per-step WMMA GEMMs (M=16 padded
//   from B=8; weights read exactly once per step = bandwidth floor) +
//   fp32 pointwise kernels (sigmoid/tanh in fp32).
// All matrix math uses v_wmma_f32_16x16x32_bf16 (wave32, 16x16 tiles).
// ---------------------------------------------------------------------------

#define B_   8
#define T_   128
#define E_   30522
#define Ep_  30528      // E padded to x32
#define H_   1000
#define Hp_  1024       // H padded to x32 (K) / x16 (N)
#define S_   64
#define NZ_  10000
#define NZp_ 10016
#define G4H_ 4000       // 4*H (already x16)

typedef __attribute__((ext_vector_type(8)))  unsigned short u16x8;
typedef __attribute__((ext_vector_type(16))) __bf16         bf16x16;
typedef __attribute__((ext_vector_type(8)))  float          f32x8;

union Frag {
  u16x8          h[2];
  unsigned short u[16];
  bf16x16        v;
};

static __device__ __forceinline__ unsigned short f2bf(float f) {
  unsigned int u = __builtin_bit_cast(unsigned int, f);
  u += 0x7FFFu + ((u >> 16) & 1u);               // round-to-nearest-even
  return (unsigned short)(u >> 16);
}
static __device__ __forceinline__ float sigmoidf_(float x) {
  return 1.0f / (1.0f + __expf(-x));
}

// ---------------------------------------------------------------------------
// GEMM (bf16 padded operands):  C[M,N] = A[M,K] @ W[N,K]^T (+ bias[N])
// One wave per 16x16 tile. K multiple of 32. lda/ldb multiples of 16.
// A-frag: lane<16 -> row m=lane, K chunks {0..7,16..23}; lane>=16 -> m=lane-16,
// K chunks {8..15,24..31}. B-frag identical pattern over rows of W (= cols N).
// Used for the M=16 sequential GEMMs (weights streamed exactly once).
// ---------------------------------------------------------------------------
__global__ void gemm_bf16_kernel(const unsigned short* __restrict__ A, int lda,
                                 const unsigned short* __restrict__ B, int ldb,
                                 float* __restrict__ C, int ldc,
                                 int Kp, const float* __restrict__ bias,
                                 int tiles_n, int total_tiles) {
  int wave = (int)((blockIdx.x * blockDim.x + threadIdx.x) >> 5);
  if (wave >= total_tiles) return;
  int tm   = wave / tiles_n;
  int tn   = wave % tiles_n;
  int lane = threadIdx.x & 31;
  int half = lane >> 4;       // 0 or 1
  int lrow = lane & 15;
  int base_k = half * 8;

  const unsigned short* ap = A + (size_t)(tm * 16 + lrow) * lda + base_k;
  const unsigned short* bp = B + (size_t)(tn * 16 + lrow) * ldb + base_k;

  f32x8 acc = {};
  for (int k0 = 0; k0 < Kp; k0 += 32) {
    if (k0 + 32 < Kp) {                    // gfx1250 global_prefetch path
      __builtin_prefetch(ap + k0 + 32, 0, 3);
      __builtin_prefetch(bp + k0 + 32, 0, 3);
    }
    Frag fa, fb;
    fa.h[0] = *(const u16x8*)(ap + k0);
    fa.h[1] = *(const u16x8*)(ap + k0 + 16);
    fb.h[0] = *(const u16x8*)(bp + k0);
    fb.h[1] = *(const u16x8*)(bp + k0 + 16);
    acc = __builtin_amdgcn_wmma_f32_16x16x32_bf16(
        false, fa.v, false, fb.v, (short)0, acc, false, false);
  }

  int row0 = tm * 16 + half * 8;
  int col  = tn * 16 + lrow;
  float bv = bias ? bias[col] : 0.0f;
#pragma unroll
  for (int r = 0; r < 8; ++r)
    C[(size_t)(row0 + r) * ldc + col] = acc[r] + bv;
}

// ---------------------------------------------------------------------------
// Big-GEMM kernel reading fp32 sources directly (bf16 convert on the fly,
// bounds-guarded).  4x M-register-blocking: each wave owns a 64x16 C block
// (4 accumulators); one B fragment per K-step is reused by 4 WMMAs, cutting
// re-streaming of the large fp32 weight operand 4x.
// amode==1 folds the [B,T,E]->[T*B,E] transpose into A addressing.
// M must be a multiple of 64 at the call site (it is: 1024 / 512).
// ---------------------------------------------------------------------------
#define MBLK 4
__global__ void gemm_f32src_kernel(const float* __restrict__ A, int lda, int amode, int Mact,
                                   const float* __restrict__ B, int ldb, int bcol0, int Nact,
                                   int Kact, int Kp,
                                   float* __restrict__ C, int ldc,
                                   const float* __restrict__ bias,
                                   int tiles_n, int total_tiles) {
  int wave = (int)((blockIdx.x * blockDim.x + threadIdx.x) >> 5);
  if (wave >= total_tiles) return;
  int tmb  = wave / tiles_n;          // 64-row M block
  int tn   = wave % tiles_n;
  int lane = threadIdx.x & 31;
  int half = lane >> 4;
  int lrow = lane & 15;
  int base_k = half * 8;

  size_t aoff[MBLK];
  bool   avalid[MBLK];
#pragma unroll
  for (int i = 0; i < MBLK; ++i) {
    int am_row = tmb * (16 * MBLK) + i * 16 + lrow;
    avalid[i]  = am_row < Mact;
    aoff[i] = amode ? (size_t)((am_row & 7) * T_ + (am_row >> 3)) * (size_t)lda
                    : (size_t)am_row * (size_t)lda;
  }
  int  bn_row = tn * 16 + lrow;
  bool bvalid = bn_row < Nact;
  size_t boff = (size_t)bn_row * (size_t)ldb + (size_t)bcol0;

  f32x8 acc[MBLK] = {};
  for (int k0 = 0; k0 < Kp; k0 += 32) {
    Frag fb;
#pragma unroll
    for (int e = 0; e < 16; ++e) {
      int k = k0 + base_k + ((e < 8) ? e : (8 + e));   // {0..7} then {16..23}
      fb.u[e] = (bvalid && k < Kact) ? f2bf(B[boff + k]) : (unsigned short)0;
    }
#pragma unroll
    for (int i = 0; i < MBLK; ++i) {
      Frag fa;
#pragma unroll
      for (int e = 0; e < 16; ++e) {
        int k = k0 + base_k + ((e < 8) ? e : (8 + e));
        fa.u[e] = (avalid[i] && k < Kact) ? f2bf(A[aoff[i] + k]) : (unsigned short)0;
      }
      acc[i] = __builtin_amdgcn_wmma_f32_16x16x32_bf16(
          false, fa.v, false, fb.v, (short)0, acc[i], false, false);
    }
  }

  int col  = tn * 16 + lrow;
  float bv = bias ? bias[col] : 0.0f;
#pragma unroll
  for (int i = 0; i < MBLK; ++i) {
    int row0 = tmb * (16 * MBLK) + i * 16 + half * 8;
#pragma unroll
    for (int r = 0; r < 8; ++r)
      C[(size_t)(row0 + r) * ldc + col] = acc[i][r] + bv;
  }
}

// ---------------------------------------------------------------------------
// Weight conversion: dst bf16 region [Rp x Creg] at (0, dst_col0) of a
// dst_ld-wide buffer, sourced from fp32 src[(r, src_col0+c)], zero-padded.
// ---------------------------------------------------------------------------
__global__ void cvt_pad_kernel(unsigned short* __restrict__ dst, int dst_ld, int dst_col0,
                               const float* __restrict__ src, int src_ld, int src_col0,
                               int Rsrc, int Cvalid, int Rp, int Creg) {
  int idx = blockIdx.x * blockDim.x + threadIdx.x;
  if (idx >= Rp * Creg) return;
  int r = idx / Creg, c = idx % Creg;
  float v = (r < Rsrc && c < Cvalid) ? src[(size_t)r * src_ld + src_col0 + c] : 0.0f;
  dst[(size_t)r * dst_ld + dst_col0 + c] = f2bf(v);
}

__global__ void bias_pad_kernel(float* dst, int n_dst, const float* src, int n_src) {
  int i = blockIdx.x * blockDim.x + threadIdx.x;
  if (i < n_dst) dst[i] = (i < n_src) ? src[i] : 0.0f;
}
__global__ void bias_sum_kernel(float* dst, const float* a, const float* b, int n) {
  int i = blockIdx.x * blockDim.x + threadIdx.x;
  if (i < n) dst[i] = a[i] + b[i];
}
__global__ void zero_f32_kernel(float* p, int n) {
  int i = blockIdx.x * blockDim.x + threadIdx.x;
  if (i < n) p[i] = 0.0f;
}
__global__ void zero_u16_kernel(unsigned short* p, int n) {
  int i = blockIdx.x * blockDim.x + threadIdx.x;
  if (i < n) p[i] = 0;
}

// init decoder input inp_bf[16, Ep] from init_dec_input [8, E]
__global__ void init_inp_kernel(const float* __restrict__ src, unsigned short* __restrict__ inp_bf) {
  int idx = blockIdx.x * blockDim.x + threadIdx.x;
  if (idx >= 16 * Ep_) return;
  int b = idx / Ep_, e = idx % Ep_;
  float v = (b < B_ && e < E_) ? src[(size_t)b * E_ + e] : 0.0f;
  inp_bf[idx] = f2bf(v);
}

// LSTM pointwise: gates = g1(optional pre-computed input part) + g2 (hh GEMM,
// bias already folded into one of them). PyTorch gate order i,f,g,o.
__global__ void lstm_pointwise_kernel(const float* __restrict__ g1, int g1_ld,
                                      const float* __restrict__ g2, int g2_ld,
                                      float* __restrict__ h, float* __restrict__ c,
                                      unsigned short* __restrict__ hbf,
                                      float* __restrict__ enc_out_t) {
  int idx = blockIdx.x * blockDim.x + threadIdx.x;
  if (idx >= B_ * Hp_) return;
  int b = idx >> 10, j = idx & (Hp_ - 1);
  if (j < H_) {
    float gi = g2[(size_t)b * g2_ld + j];
    float gf = g2[(size_t)b * g2_ld + H_ + j];
    float gg = g2[(size_t)b * g2_ld + 2 * H_ + j];
    float go = g2[(size_t)b * g2_ld + 3 * H_ + j];
    if (g1) {
      gi += g1[(size_t)b * g1_ld + j];
      gf += g1[(size_t)b * g1_ld + H_ + j];
      gg += g1[(size_t)b * g1_ld + 2 * H_ + j];
      go += g1[(size_t)b * g1_ld + 3 * H_ + j];
    }
    float i = sigmoidf_(gi), f = sigmoidf_(gf), g = tanhf(gg), o = sigmoidf_(go);
    float cn = f * c[idx] + i * g;
    float hn = o * tanhf(cn);
    c[idx] = cn;
    h[idx] = hn;
    hbf[idx] = f2bf(hn);
    if (enc_out_t) enc_out_t[(size_t)b * H_ + j] = hn;
  } else {
    c[idx] = 0.0f; h[idx] = 0.0f; hbf[idx] = 0;
  }
}

// emb = embC (inp part + bias) + nemb_s (precomputed noise part), zero-padded
__global__ void emb_finalize_kernel(const float* __restrict__ embC,
                                    const float* __restrict__ nemb_s,
                                    float* __restrict__ emb) {
  int idx = blockIdx.x * blockDim.x + threadIdx.x;
  if (idx >= B_ * Hp_) return;
  int j = idx & (Hp_ - 1);
  emb[idx] = (j < H_) ? (embC[idx] + nemb_s[idx]) : 0.0f;
}

// attn_in[16,3072] rows 0..7: [emb | h | noise2_s] (segments padded to 1024)
__global__ void build_attn_in_kernel(const float* __restrict__ emb,
                                     const float* __restrict__ h,
                                     const float* __restrict__ n2s,
                                     unsigned short* __restrict__ dst) {
  int idx = blockIdx.x * blockDim.x + threadIdx.x;
  if (idx >= B_ * 3072) return;
  int b = idx / 3072, col = idx % 3072;
  float v;
  if (col < Hp_)            v = emb[b * Hp_ + col];
  else if (col < 2 * Hp_)   v = h[b * Hp_ + (col - Hp_)];
  else { int j = col - 2 * Hp_; v = (j < H_) ? n2s[(size_t)b * H_ + j] : 0.0f; }
  dst[(size_t)b * 3072 + col] = f2bf(v);
}

__global__ void softmax_kernel(const float* __restrict__ scores, int ld,
                               float* __restrict__ aw) {
  __shared__ float red[T_];
  int b = blockIdx.x, t = threadIdx.x;
  float v = scores[(size_t)b * ld + t];
  red[t] = v; __syncthreads();
  for (int s = T_ / 2; s > 0; s >>= 1) {
    if (t < s) red[t] = fmaxf(red[t], red[t + s]);
    __syncthreads();
  }
  float m = red[0]; __syncthreads();
  float e = __expf(v - m);
  red[t] = e; __syncthreads();
  for (int s = T_ / 2; s > 0; s >>= 1) {
    if (t < s) red[t] += red[t + s];
    __syncthreads();
  }
  aw[b * T_ + t] = e / red[0];
}

// applied[b,j] = sum_t aw[b,t] * enc_out[t,b,j]
__global__ void applied_kernel(const float* __restrict__ aw,
                               const float* __restrict__ enc_out,
                               float* __restrict__ applied) {
  int idx = blockIdx.x * blockDim.x + threadIdx.x;
  if (idx >= B_ * Hp_) return;
  int b = idx >> 10, j = idx & (Hp_ - 1);
  float acc = 0.0f;
  if (j < H_) {
    for (int t = 0; t < T_; ++t)
      acc += aw[b * T_ + t] * enc_out[(size_t)(t * B_ + b) * H_ + j];
  }
  applied[idx] = acc;
}

// comb_in[16,2048] rows 0..7: [emb | applied]
__global__ void build_comb_in_kernel(const float* __restrict__ emb,
                                     const float* __restrict__ applied,
                                     unsigned short* __restrict__ dst) {
  int idx = blockIdx.x * blockDim.x + threadIdx.x;
  if (idx >= B_ * 2048) return;
  int b = idx / 2048, col = idx % 2048;
  float v = (col < Hp_) ? emb[b * Hp_ + col] : applied[b * Hp_ + (col - Hp_)];
  dst[(size_t)b * 2048 + col] = f2bf(v);
}

// dec_in[16,2048] rows 0..7: cols 0..1023 = relu(combC), cols 1024.. = h
__global__ void relu_to_decin_kernel(const float* __restrict__ combC,
                                     const float* __restrict__ h,
                                     unsigned short* __restrict__ dec_in) {
  int idx = blockIdx.x * blockDim.x + threadIdx.x;
  if (idx >= B_ * 2048) return;
  int b = idx / 2048, col = idx % 2048;
  float v;
  if (col < Hp_) { int j = col; v = (j < H_) ? fmaxf(combC[b * Hp_ + j], 0.0f) : 0.0f; }
  else           v = h[b * Hp_ + (col - Hp_)];
  dec_in[(size_t)b * 2048 + col] = f2bf(v);
}

// write summary[s] from outC and refresh inp_bf for next step
__global__ void out_finalize_kernel(const float* __restrict__ outC,
                                    float* __restrict__ out_s,
                                    unsigned short* __restrict__ inp_bf) {
  int idx = blockIdx.x * blockDim.x + threadIdx.x;
  if (idx >= B_ * Ep_) return;
  int b = idx / Ep_, e = idx % Ep_;
  float v = 0.0f;
  if (e < E_) {
    v = outC[(size_t)b * Ep_ + e];
    out_s[(size_t)b * E_ + e] = v;
  }
  inp_bf[(size_t)b * Ep_ + e] = f2bf(v);
}

// ---------------------------------------------------------------------------
extern "C" void kernel_launch(void* const* d_in, const int* in_sizes, int n_in,
                              void* d_out, int out_size, void* d_ws, size_t ws_size,
                              hipStream_t stream) {
  const float* input    = (const float*)d_in[0];
  const float* noise1   = (const float*)d_in[2];
  const float* noise2   = (const float*)d_in[3];
  const float* init_dec = (const float*)d_in[4];
  const float* enc_Wih  = (const float*)d_in[5];
  const float* enc_Whh  = (const float*)d_in[6];
  const float* enc_bih  = (const float*)d_in[7];
  const float* enc_bhh  = (const float*)d_in[8];
  const float* dec_Wih  = (const float*)d_in[9];
  const float* dec_Whh  = (const float*)d_in[10];
  const float* dec_bih  = (const float*)d_in[11];
  const float* dec_bhh  = (const float*)d_in[12];
  const float* emb_W    = (const float*)d_in[13];
  const float* emb_b    = (const float*)d_in[14];
  const float* attn_W   = (const float*)d_in[15];
  const float* attn_b   = (const float*)d_in[16];
  const float* comb_W   = (const float*)d_in[17];
  const float* comb_b   = (const float*)d_in[18];
  const float* out_W    = (const float*)d_in[19];
  const float* out_b    = (const float*)d_in[20];
  float* out = (float*)d_out;

  // ---- workspace bump allocator (total ~182 MB) ----
  char* base = (char*)d_ws;
  size_t off = 0;
  auto alloc = [&](size_t bytes) -> void* {
    void* r = base + off;
    off = (off + bytes + 255) & ~(size_t)255;
    return r;
  };
  unsigned short* embE_bf   = (unsigned short*)alloc((size_t)Hp_ * Ep_ * 2);
  unsigned short* outW_bf   = (unsigned short*)alloc((size_t)Ep_ * Hp_ * 2);
  unsigned short* encWhh_bf = (unsigned short*)alloc((size_t)G4H_ * Hp_ * 2);
  unsigned short* decW_bf   = (unsigned short*)alloc((size_t)G4H_ * 2048 * 2);
  unsigned short* attnW_bf  = (unsigned short*)alloc((size_t)T_ * 3072 * 2);
  unsigned short* combW_bf  = (unsigned short*)alloc((size_t)Hp_ * 2048 * 2);
  float* encX    = (float*)alloc((size_t)T_ * B_ * G4H_ * 4);
  float* nemb    = (float*)alloc((size_t)S_ * B_ * Hp_ * 4);
  float* enc_out = (float*)alloc((size_t)T_ * B_ * H_ * 4);
  float* h_buf   = (float*)alloc((size_t)B_ * Hp_ * 4);
  float* c_buf   = (float*)alloc((size_t)B_ * Hp_ * 4);
  unsigned short* hbf     = (unsigned short*)alloc((size_t)16 * Hp_ * 2);
  unsigned short* inp_bf  = (unsigned short*)alloc((size_t)16 * Ep_ * 2);
  float* gates   = (float*)alloc((size_t)16 * G4H_ * 4);
  float* embC    = (float*)alloc((size_t)16 * Hp_ * 4);
  float* emb     = (float*)alloc((size_t)B_ * Hp_ * 4);
  unsigned short* attn_in = (unsigned short*)alloc((size_t)16 * 3072 * 2);
  float* scores  = (float*)alloc((size_t)16 * T_ * 4);
  float* aw      = (float*)alloc((size_t)B_ * T_ * 4);
  float* applied = (float*)alloc((size_t)B_ * Hp_ * 4);
  unsigned short* comb_in = (unsigned short*)alloc((size_t)16 * 2048 * 2);
  float* combC   = (float*)alloc((size_t)16 * Hp_ * 4);
  unsigned short* dec_in  = (unsigned short*)alloc((size_t)16 * 2048 * 2);
  float* outC    = (float*)alloc((size_t)16 * Ep_ * 4);
  float* enc_bsum = (float*)alloc(G4H_ * 4);
  float* dec_bsum = (float*)alloc(G4H_ * 4);
  float* emb_bp   = (float*)alloc(Hp_ * 4);
  float* comb_bp  = (float*)alloc(Hp_ * 4);
  float* out_bp   = (float*)alloc(Ep_ * 4);

#define GRID1(n) ((n) + 255) / 256, 256, 0, stream
#define GEMMGRID(tiles) ((tiles) + 7) / 8, 256, 0, stream   // 8 waves / block

  // ---- phase 0: bias prep + weight conversion + state init ----
  bias_sum_kernel<<<GRID1(G4H_)>>>(enc_bsum, enc_bih, enc_bhh, G4H_);
  bias_sum_kernel<<<GRID1(G4H_)>>>(dec_bsum, dec_bih, dec_bhh, G4H_);
  bias_pad_kernel<<<GRID1(Hp_)>>>(emb_bp, Hp_, emb_b, H_);
  bias_pad_kernel<<<GRID1(Hp_)>>>(comb_bp, Hp_, comb_b, H_);
  bias_pad_kernel<<<GRID1(Ep_)>>>(out_bp, Ep_, out_b, E_);

  cvt_pad_kernel<<<GRID1(Hp_ * Ep_)>>>(embE_bf, Ep_, 0, emb_W, E_ + NZ_, 0, H_, E_, Hp_, Ep_);
  cvt_pad_kernel<<<GRID1(Ep_ * Hp_)>>>(outW_bf, Hp_, 0, out_W, H_, 0, E_, H_, Ep_, Hp_);
  cvt_pad_kernel<<<GRID1(G4H_ * Hp_)>>>(encWhh_bf, Hp_, 0, enc_Whh, H_, 0, G4H_, H_, G4H_, Hp_);
  cvt_pad_kernel<<<GRID1(G4H_ * Hp_)>>>(decW_bf, 2048, 0,   dec_Wih, H_, 0, G4H_, H_, G4H_, Hp_);
  cvt_pad_kernel<<<GRID1(G4H_ * Hp_)>>>(decW_bf, 2048, Hp_, dec_Whh, H_, 0, G4H_, H_, G4H_, Hp_);
  cvt_pad_kernel<<<GRID1(T_ * Hp_)>>>(attnW_bf, 3072, 0,        attn_W, 3 * H_, 0,      T_, H_, T_, Hp_);
  cvt_pad_kernel<<<GRID1(T_ * Hp_)>>>(attnW_bf, 3072, Hp_,      attn_W, 3 * H_, H_,     T_, H_, T_, Hp_);
  cvt_pad_kernel<<<GRID1(T_ * Hp_)>>>(attnW_bf, 3072, 2 * Hp_,  attn_W, 3 * H_, 2 * H_, T_, H_, T_, Hp_);
  cvt_pad_kernel<<<GRID1(Hp_ * Hp_)>>>(combW_bf, 2048, 0,   comb_W, 2 * H_, 0,  H_, H_, Hp_, Hp_);
  cvt_pad_kernel<<<GRID1(Hp_ * Hp_)>>>(combW_bf, 2048, Hp_, comb_W, 2 * H_, H_, H_, H_, Hp_, Hp_);

  zero_f32_kernel<<<GRID1(B_ * Hp_)>>>(h_buf, B_ * Hp_);
  zero_f32_kernel<<<GRID1(B_ * Hp_)>>>(c_buf, B_ * Hp_);
  zero_u16_kernel<<<GRID1(16 * Hp_)>>>(hbf, 16 * Hp_);
  zero_u16_kernel<<<GRID1(16 * 3072)>>>(attn_in, 16 * 3072);
  zero_u16_kernel<<<GRID1(16 * 2048)>>>(comb_in, 16 * 2048);
  zero_u16_kernel<<<GRID1(16 * 2048)>>>(dec_in, 16 * 2048);
  init_inp_kernel<<<GRID1(16 * Ep_)>>>(init_dec, inp_bf);

  // ---- phase 0b: hoisted big GEMMs (fp32 sources, on-the-fly bf16) ----
  // encX[T*B, 4H] = x^T @ enc_Wih^T + (bih+bhh)    (transpose folded: amode=1)
  {
    int tiles = (T_ * B_ / (16 * MBLK)) * (G4H_ / 16);   // 16 * 250 = 4000
    gemm_f32src_kernel<<<GEMMGRID(tiles)>>>(input, E_, 1, T_ * B_,
                                            enc_Wih, E_, 0, G4H_,
                                            E_, Ep_, encX, G4H_, enc_bsum,
                                            G4H_ / 16, tiles);
  }
  // nemb[S*B, Hp] = noise1 @ emb_W[:, E:]^T
  {
    int tiles = (S_ * B_ / (16 * MBLK)) * (Hp_ / 16);    // 8 * 64 = 512
    gemm_f32src_kernel<<<GEMMGRID(tiles)>>>(noise1, NZ_, 0, S_ * B_,
                                            emb_W, E_ + NZ_, E_, H_,
                                            NZ_, NZp_, nemb, Hp_, nullptr,
                                            Hp_ / 16, tiles);
  }

  // ---- phase 1: encoder recurrence ----
  for (int t = 0; t < T_; ++t) {
    gemm_bf16_kernel<<<GEMMGRID(G4H_ / 16)>>>(hbf, Hp_, encWhh_bf, Hp_,
                                              gates, G4H_, Hp_, nullptr,
                                              G4H_ / 16, G4H_ / 16);
    lstm_pointwise_kernel<<<GRID1(B_ * Hp_)>>>(encX + (size_t)t * B_ * G4H_, G4H_,
                                               gates, G4H_, h_buf, c_buf, hbf,
                                               enc_out + (size_t)t * B_ * H_);
  }

  // ---- phase 2: attention decoder ----
  for (int s = 0; s < S_; ++s) {
    // emb input part: inp_bf @ embE^T + emb_b
    gemm_bf16_kernel<<<GEMMGRID(Hp_ / 16)>>>(inp_bf, Ep_, embE_bf, Ep_,
                                             embC, Hp_, Ep_, emb_bp,
                                             Hp_ / 16, Hp_ / 16);
    emb_finalize_kernel<<<GRID1(B_ * Hp_)>>>(embC, nemb + (size_t)s * B_ * Hp_, emb);
    build_attn_in_kernel<<<GRID1(B_ * 3072)>>>(emb, h_buf,
                                               noise2 + (size_t)s * B_ * H_, attn_in);
    gemm_bf16_kernel<<<GEMMGRID(T_ / 16)>>>(attn_in, 3072, attnW_bf, 3072,
                                            scores, T_, 3072, attn_b,
                                            T_ / 16, T_ / 16);
    softmax_kernel<<<B_, T_, 0, stream>>>(scores, T_, aw);
    applied_kernel<<<GRID1(B_ * Hp_)>>>(aw, enc_out, applied);
    build_comb_in_kernel<<<GRID1(B_ * 2048)>>>(emb, applied, comb_in);
    gemm_bf16_kernel<<<GEMMGRID(Hp_ / 16)>>>(comb_in, 2048, combW_bf, 2048,
                                             combC, Hp_, 2048, comb_bp,
                                             Hp_ / 16, Hp_ / 16);
    relu_to_decin_kernel<<<GRID1(B_ * 2048)>>>(combC, h_buf, dec_in);
    // decoder LSTM gates: [out | h] @ [dec_Wih | dec_Whh]^T + (bih+bhh)
    gemm_bf16_kernel<<<GEMMGRID(G4H_ / 16)>>>(dec_in, 2048, decW_bf, 2048,
                                              gates, G4H_, 2048, dec_bsum,
                                              G4H_ / 16, G4H_ / 16);
    lstm_pointwise_kernel<<<GRID1(B_ * Hp_)>>>(nullptr, 0, gates, G4H_,
                                               h_buf, c_buf, hbf, nullptr);
    // inp = h @ out_W^T + out_b  -> summary[s], next-step inp_bf
    gemm_bf16_kernel<<<GEMMGRID(Ep_ / 16)>>>(hbf, Hp_, outW_bf, Hp_,
                                             outC, Ep_, Hp_, out_bp,
                                             Ep_ / 16, Ep_ / 16);
    out_finalize_kernel<<<GRID1(B_ * Ep_)>>>(outC, out + (size_t)s * B_ * E_, inp_bf);
  }

#undef GRID1
#undef GEMMGRID
}
